// MetaLSTMCell_87144886436136
// MI455X (gfx1250) — matrix operand order repaired and checked by reading, not compiled, not run on hardware
//
#include <hip/hip_runtime.h>

// ---------------------------------------------------------------------------
// HyperLSTM (MetaLSTM) cell for MI455X / gfx1250, bf16 WMMA with f32 accum.
// B=8192, I=512, H=512, HH=128, E=32.
// Main + meta GEMMs: LDS-staged, double-buffered weight tiles (8x cross-wave
// reuse), register-pipelined A fragments, fused LSTM gate epilogues.
// ---------------------------------------------------------------------------

#define Bz   8192
#define Iz   512
#define Hz   512
#define HHz  128
#define Ez   32

// LDS tile geometry: 16x32 bf16 tiles, rows padded to 40 halfwords (80B)
// for bank spread, 16B aligned.
#define TROW    40
#define TILEHW  (16 * TROW)      // 640 halfwords per tile
#define BUFHW   (8 * TILEHW)     // main kernel: 8 tiles / buffer
#define MBUFHW  (4 * TILEHW)     // meta kernel: 4 tiles / buffer

typedef __attribute__((ext_vector_type(16))) __bf16 v16bf;
typedef __attribute__((ext_vector_type(8)))  float  v8f;

union V8 { float f[8]; v8f v; };
union VB { unsigned int u[8]; v16bf v; };

__device__ __forceinline__ unsigned short f2bf(float f) {
    unsigned int u = __float_as_uint(f);
    unsigned int r = u + 0x7FFFu + ((u >> 16) & 1u);   // round-to-nearest-even
    return (unsigned short)(r >> 16);
}

__device__ __forceinline__ float sigmoidf(float x) {
    return 1.0f / (1.0f + __expf(-x));
}

__device__ __forceinline__ v8f wmma_bf16(v16bf a, v16bf b, v8f c) {
    return __builtin_amdgcn_wmma_f32_16x16x32_bf16(
        /*neg_a=*/false, a, /*neg_b=*/false, b,
        /*c_mod=*/(short)0, c, /*reuse_a=*/false, /*reuse_b=*/false);
}

// A fragment: 16x32 bf16 tile from row-major [*, ld] at (m0, k0).
// lane r=lane&15 -> row m0+r; half=lane>>4 selects K {0..15} vs {16..31}.
__device__ __forceinline__ v16bf load_a_frag(const unsigned short* __restrict__ p,
                                             int m0, int k0, int ld, int r, int half) {
    VB t;
    const unsigned int* q = (const unsigned int*)(p + (size_t)(m0 + r) * ld + k0);
    int base = half * 8;
#pragma unroll
    for (int j = 0; j < 4; ++j) t.u[j]     = q[base + j];
#pragma unroll
    for (int j = 0; j < 4; ++j) t.u[4 + j] = q[4 + base + j];
    return t.v;
}

// B fragment from global: 32x16 tile, B[k,n] = W[n,k], W row-major [*, ld].
__device__ __forceinline__ v16bf load_b_frag(const unsigned short* __restrict__ w,
                                             int n0, int k0, int ld, int r, int half) {
    VB t;
    const unsigned int* q = (const unsigned int*)(w + (size_t)(n0 + r) * ld + k0 + half * 16);
#pragma unroll
    for (int j = 0; j < 8; ++j) t.u[j] = q[j];
    return t.v;
}

// B fragment from an LDS-staged tile (16 rows x 32 k, row stride TROW halfwords).
__device__ __forceinline__ v16bf lds_b_frag(const unsigned short* buf, int t,
                                            int r, int half) {
    VB x;
    const unsigned int* q = (const unsigned int*)(buf + t * TILEHW + r * TROW + half * 16);
#pragma unroll
    for (int j = 0; j < 8; ++j) x.u[j] = q[j];
    return x.v;
}

// Cooperative stage of 8 weight tiles (4 gates x {WiH,WHH}) for one k-step.
// 256 threads, each copies 16 halfwords (32B) as 2 x b128.
__device__ __forceinline__ void stage_B8(unsigned short* dst,
                                         const unsigned short* __restrict__ wiH,
                                         const unsigned short* __restrict__ wHH,
                                         int h0, int k0, int tid) {
    int f   = tid * 16;        // 0..4095 halfwords of payload
    int t   = f >> 9;          // tile 0..7
    int rem = f & 511;
    int rr  = rem >> 5;        // row 0..15
    int kk  = rem & 31;        // 0 or 16
    int g   = t >> 1;
    const unsigned short* W = (t & 1) ? wHH : wiH;
    const uint4* s = (const uint4*)(W + (size_t)(g * Hz + h0 + rr) * Iz + k0 + kk);
    uint4* d = (uint4*)(dst + t * TILEHW + rr * TROW + kk);
    d[0] = s[0];
    d[1] = s[1];
}

// Cooperative stage of 4 gate tiles from one weight matrix (meta kernel).
// 256 threads, each copies 8 halfwords (16B) as 1 x b128.
__device__ __forceinline__ void stage_B4(unsigned short* dst,
                                         const unsigned short* __restrict__ W,
                                         int ld, int h0, int kcol, int tid) {
    int f   = tid * 8;         // 0..2047 halfwords of payload
    int t   = f >> 9;          // tile (gate) 0..3
    int rem = f & 511;
    int rr  = rem >> 5;        // row 0..15
    int kk  = rem & 31;        // 0,8,16,24
    const uint4* s = (const uint4*)(W + (size_t)(t * HHz + h0 + rr) * ld + kcol + kk);
    uint4* d = (uint4*)(dst + t * TILEHW + rr * TROW + kk);
    d[0] = s[0];
}

// ---------------------------------------------------------------------------
// f32 -> bf16 conversion (4 elements / thread)
// ---------------------------------------------------------------------------
__global__ void cvt_f32_bf16(const float* __restrict__ src,
                             unsigned short* __restrict__ dst, int n4) {
    int i = blockIdx.x * blockDim.x + threadIdx.x;
    if (i < n4) {
        float4 v = ((const float4*)src)[i];
        ushort4 o;
        o.x = f2bf(v.x); o.y = f2bf(v.y); o.z = f2bf(v.z); o.w = f2bf(v.w);
        ((ushort4*)dst)[i] = o;
    }
}

// ---------------------------------------------------------------------------
// z-embeddings: z[s] = meta_h @ W_hz[s].T (+ bias), stored bf16.  [B, E]
// ---------------------------------------------------------------------------
__global__ void z_gemm(const unsigned short* __restrict__ mtb,
                       const unsigned short* __restrict__ whzi,
                       const unsigned short* __restrict__ whzH,
                       const unsigned short* __restrict__ whzb,
                       const float* __restrict__ bias_i,
                       const float* __restrict__ bias_H,
                       unsigned short* __restrict__ zib,
                       unsigned short* __restrict__ zHb,
                       unsigned short* __restrict__ zbb) {
    int gw   = blockIdx.x * 8 + (threadIdx.x >> 5);
    int lane = threadIdx.x & 31;
    int r    = lane & 15, half = lane >> 4;

    int mtile = gw / 6;
    int rem   = gw % 6;
    int s     = rem >> 1;
    int n0    = (rem & 1) * 16;
    int m0    = mtile * 16;

    const unsigned short* W = (s == 0) ? whzi : (s == 1) ? whzH : whzb;
    unsigned short*       Z = (s == 0) ? zib  : (s == 1) ? zHb  : zbb;
    float bv = (s == 0) ? bias_i[n0 + r] : (s == 1) ? bias_H[n0 + r] : 0.0f;

    v8f acc = {};
#pragma unroll
    for (int k0 = 0; k0 < HHz; k0 += 32) {
        v16bf a = load_a_frag(mtb, m0, k0, HHz, r, half);
        v16bf b = load_b_frag(W,   n0, k0, HHz, r, half);
        acc = wmma_bf16(a, b, acc);
    }
    V8 A; A.v = acc;
#pragma unroll
    for (int j = 0; j < 8; ++j) {
        int row = m0 + half * 8 + j;
        Z[(size_t)row * Ez + n0 + r] = f2bf(A.f[j] + bv);
    }
}

// ---------------------------------------------------------------------------
// Meta (hyper) LSTM: pre = [input|main_h]@W_ih.T + meta_h@W_hh.T + bias_hyper
// fused with gate epilogue -> meta_h_new, meta_c_new.
// Block = 8 waves = 128 rows x one 16-col h-tile; weight tiles shared by all
// waves via double-buffered LDS; 36 unified k-steps (16 input, 16 main_h,
// 4 meta_h) with wave-uniform segment selection.
// ---------------------------------------------------------------------------
__global__ void __launch_bounds__(256)
meta_gemm_lstm(const unsigned short* __restrict__ inb,
               const unsigned short* __restrict__ mhb,
               const unsigned short* __restrict__ mtb,
               const unsigned short* __restrict__ wih,   // [4*HH, I+H]
               const unsigned short* __restrict__ whh,   // [4*HH, HH]
               const float* __restrict__ bias_hyper,
               const float* __restrict__ meta_c,
               float* __restrict__ out_h,
               float* __restrict__ out_c) {
    __shared__ unsigned short lds[2 * MBUFHW];

    int tid  = threadIdx.x;
    int wave = tid >> 5;
    int lane = tid & 31;
    int r    = lane & 15, half = lane >> 4;

    int blkm = blockIdx.x >> 3;          // 64 m-blocks of 128 rows
    int ht   = blockIdx.x & 7;           // 8 h-tiles (HH/16)
    int h0   = ht * 16;
    int m0   = (blkm * 8 + wave) * 16;

    auto stageM = [&](unsigned short* dst, int ks) {
        if (ks < 32) stage_B4(dst, wih, Iz + Hz, h0, ks * 32, tid);
        else         stage_B4(dst, whh, HHz,     h0, (ks - 32) * 32, tid);
    };
    auto loadA = [&](int ks) -> v16bf {
        if (ks < 16)      return load_a_frag(inb, m0, ks * 32, Iz, r, half);
        else if (ks < 32) return load_a_frag(mhb, m0, (ks - 16) * 32, Hz, r, half);
        else              return load_a_frag(mtb, m0, (ks - 32) * 32, HHz, r, half);
    };

    stageM(lds, 0);
    v8f acc[4] = {{}, {}, {}, {}};
    v16bf a = loadA(0);

    for (int ks = 0; ks < 36; ++ks) {
        __syncthreads();                              // buf[ks&1] ready for all
        const unsigned short* cur = lds + (ks & 1) * MBUFHW;

        v16bf an = a;
        if (ks < 35) {
            stageM(lds + ((ks + 1) & 1) * MBUFHW, ks + 1);
            an = loadA(ks + 1);
        }
#pragma unroll
        for (int g = 0; g < 4; ++g) {
            v16bf b = lds_b_frag(cur, g, r, half);
            acc[g] = wmma_bf16(a, b, acc[g]);
        }
        a = an;
    }

    V8 P[4];
#pragma unroll
    for (int g = 0; g < 4; ++g) {
        P[g].v = acc[g];
        float bn = bias_hyper[g * HHz + h0 + r];
#pragma unroll
        for (int j = 0; j < 8; ++j) P[g].f[j] += bn;
    }

    int col = h0 + r;
#pragma unroll
    for (int j = 0; j < 8; ++j) {
        int row = m0 + half * 8 + j;
        size_t idx = (size_t)row * HHz + col;
        float cprev = meta_c[idx];
        float iv = sigmoidf(P[0].f[j]);
        float fv = sigmoidf(P[1].f[j]);
        float gv = tanhf(P[2].f[j]);
        float ov = sigmoidf(P[3].f[j]);
        float cn = fv * cprev + iv * gv;
        out_c[idx] = cn;
        out_h[idx] = ov * tanhf(cn);
    }
}

// ---------------------------------------------------------------------------
// Main LSTM: pre = (zi@Wdzi.T)*(x@WiH.T) + (zH@WdzH.T)*(h@WHH.T) + zb@Wbz.T + b
// Block = 8 waves = 128 rows x one 16-col h-tile. Weight tiles double-buffered
// through LDS; A fragments software-pipelined in registers.
// ---------------------------------------------------------------------------
__global__ void __launch_bounds__(256)
main_gemm_lstm(const unsigned short* __restrict__ inb,
               const unsigned short* __restrict__ mhb,
               const unsigned short* __restrict__ wiH,   // [4H, I]
               const unsigned short* __restrict__ wHH,   // [4H, H]
               const unsigned short* __restrict__ zib,
               const unsigned short* __restrict__ zHb,
               const unsigned short* __restrict__ zbb,
               const unsigned short* __restrict__ wdzi,  // [4H, E]
               const unsigned short* __restrict__ wdzH,
               const unsigned short* __restrict__ wbz,
               const float* __restrict__ bias,           // [4H]
               const float* __restrict__ main_c,         // [B,H] f32
               float* __restrict__ out_h,
               float* __restrict__ out_c) {
    __shared__ unsigned short lds[2 * BUFHW];

    int tid  = threadIdx.x;
    int wave = tid >> 5;
    int lane = tid & 31;
    int r    = lane & 15, half = lane >> 4;

    int blkm = blockIdx.x >> 5;          // 64 m-blocks of 128 rows
    int ht   = blockIdx.x & 31;          // 32 h-tiles
    int h0   = ht * 16;
    int m0   = (blkm * 8 + wave) * 16;

    // prologue: stage k-step 0, preload A fragments for k-step 0
    stage_B8(lds, wiH, wHH, h0, 0, tid);

    v8f q1[4] = {{}, {}, {}, {}};
    v8f q2[4] = {{}, {}, {}, {}};
    v16bf aI = load_a_frag(inb, m0, 0, Iz, r, half);
    v16bf aH = load_a_frag(mhb, m0, 0, Hz, r, half);

    for (int ks = 0; ks < 16; ++ks) {
        __syncthreads();                              // buf[ks&1] ready for all
        const unsigned short* cur = lds + (ks & 1) * BUFHW;

        v16bf aIn = aI, aHn = aH;
        if (ks < 15) {
            // overlap: stage next weight tiles + prefetch next A fragments
            stage_B8(lds + ((ks + 1) & 1) * BUFHW, wiH, wHH, h0, (ks + 1) * 32, tid);
            aIn = load_a_frag(inb, m0, (ks + 1) * 32, Iz, r, half);
            aHn = load_a_frag(mhb, m0, (ks + 1) * 32, Hz, r, half);
        }

#pragma unroll
        for (int g = 0; g < 4; ++g) {
            v16bf b1 = lds_b_frag(cur, g * 2 + 0, r, half);
            q1[g] = wmma_bf16(aI, b1, q1[g]);
            v16bf b2 = lds_b_frag(cur, g * 2 + 1, r, half);
            q2[g] = wmma_bf16(aH, b2, q2[g]);
        }
        aI = aIn;
        aH = aHn;
    }

    // z modulation terms: single K=32 WMMA each; A frags shared across gates
    v16bf az1 = load_a_frag(zib, m0, 0, Ez, r, half);
    v16bf az2 = load_a_frag(zHb, m0, 0, Ez, r, half);
    v16bf az3 = load_a_frag(zbb, m0, 0, Ez, r, half);

    V8 P[4];
#pragma unroll
    for (int g = 0; g < 4; ++g) {
        int nb = g * Hz + h0;
        v8f zero = {};
        V8 p1, p2, p3, a1, a2;
        p1.v = wmma_bf16(az1, load_b_frag(wdzi, nb, 0, Ez, r, half), zero);
        p2.v = wmma_bf16(az2, load_b_frag(wdzH, nb, 0, Ez, r, half), zero);
        p3.v = wmma_bf16(az3, load_b_frag(wbz,  nb, 0, Ez, r, half), zero);
        a1.v = q1[g];
        a2.v = q2[g];
        float bn = bias[nb + r];
#pragma unroll
        for (int j = 0; j < 8; ++j)
            P[g].f[j] = p1.f[j] * a1.f[j] + p2.f[j] * a2.f[j] + p3.f[j] + bn;
    }

    int col = h0 + r;
#pragma unroll
    for (int j = 0; j < 8; ++j) {
        int row = m0 + half * 8 + j;
        size_t idx = (size_t)row * Hz + col;
        float cprev = main_c[idx];
        float iv = sigmoidf(P[0].f[j]);
        float fv = sigmoidf(P[1].f[j]);
        float gv = tanhf(P[2].f[j]);
        float ov = sigmoidf(P[3].f[j]);
        float cn = fv * cprev + iv * gv;
        out_c[idx] = cn;
        out_h[idx] = ov * tanhf(cn);
    }
}

// ---------------------------------------------------------------------------
// Launcher
// ---------------------------------------------------------------------------
static inline void launch_cvt(const float* s, unsigned short* d, size_t n, hipStream_t st) {
    int n4 = (int)(n / 4);
    cvt_f32_bf16<<<(n4 + 255) / 256, 256, 0, st>>>(s, d, n4);
}

extern "C" void kernel_launch(void* const* d_in, const int* in_sizes, int n_in,
                              void* d_out, int out_size, void* d_ws, size_t ws_size,
                              hipStream_t stream) {
    const float* input      = (const float*)d_in[0];
    const float* main_h     = (const float*)d_in[1];
    const float* main_c     = (const float*)d_in[2];
    const float* meta_h     = (const float*)d_in[3];
    const float* meta_c     = (const float*)d_in[4];
    const float* weight_iH  = (const float*)d_in[5];
    const float* weight_HH  = (const float*)d_in[6];
    const float* weight_ih  = (const float*)d_in[7];
    const float* weight_hh  = (const float*)d_in[8];
    const float* weight_hzi = (const float*)d_in[9];
    const float* weight_hzH = (const float*)d_in[10];
    const float* weight_hzb = (const float*)d_in[11];
    const float* weight_dzi = (const float*)d_in[12];
    const float* weight_dzH = (const float*)d_in[13];
    const float* weight_bz  = (const float*)d_in[14];
    const float* bias_i     = (const float*)d_in[15];
    const float* bias_H     = (const float*)d_in[16];
    const float* bias       = (const float*)d_in[17];
    const float* bias_hyper = (const float*)d_in[18];

    float* out = (float*)d_out;
    float* out_mainh = out;
    float* out_mainc = out + (size_t)Bz * Hz;
    float* out_metah = out + (size_t)2 * Bz * Hz;
    float* out_metac = out + (size_t)2 * Bz * Hz + (size_t)Bz * HHz;

    // workspace carve-up (bf16 halfwords)
    unsigned short* ws = (unsigned short*)d_ws;
    size_t off = 0;
    unsigned short* inb  = ws + off; off += (size_t)Bz * Iz;
    unsigned short* mhb  = ws + off; off += (size_t)Bz * Hz;
    unsigned short* mtb  = ws + off; off += (size_t)Bz * HHz;
    unsigned short* wiH  = ws + off; off += (size_t)4 * Hz * Iz;
    unsigned short* wHH  = ws + off; off += (size_t)4 * Hz * Hz;
    unsigned short* wih  = ws + off; off += (size_t)4 * HHz * (Iz + Hz);
    unsigned short* whh  = ws + off; off += (size_t)4 * HHz * HHz;
    unsigned short* whzi = ws + off; off += (size_t)Ez * HHz;
    unsigned short* whzH = ws + off; off += (size_t)Ez * HHz;
    unsigned short* whzb = ws + off; off += (size_t)Ez * HHz;
    unsigned short* wdzi = ws + off; off += (size_t)4 * Hz * Ez;
    unsigned short* wdzH = ws + off; off += (size_t)4 * Hz * Ez;
    unsigned short* wbz  = ws + off; off += (size_t)4 * Hz * Ez;
    unsigned short* zib  = ws + off; off += (size_t)Bz * Ez;
    unsigned short* zHb  = ws + off; off += (size_t)Bz * Ez;
    unsigned short* zbb  = ws + off; off += (size_t)Bz * Ez;
    (void)ws_size; (void)n_in; (void)in_sizes; (void)out_size;

    // f32 -> bf16 conversions
    launch_cvt(input,      inb,  (size_t)Bz * Iz,             stream);
    launch_cvt(main_h,     mhb,  (size_t)Bz * Hz,             stream);
    launch_cvt(meta_h,     mtb,  (size_t)Bz * HHz,            stream);
    launch_cvt(weight_iH,  wiH,  (size_t)4 * Hz * Iz,         stream);
    launch_cvt(weight_HH,  wHH,  (size_t)4 * Hz * Hz,         stream);
    launch_cvt(weight_ih,  wih,  (size_t)4 * HHz * (Iz + Hz), stream);
    launch_cvt(weight_hh,  whh,  (size_t)4 * HHz * HHz,       stream);
    launch_cvt(weight_hzi, whzi, (size_t)Ez * HHz,            stream);
    launch_cvt(weight_hzH, whzH, (size_t)Ez * HHz,            stream);
    launch_cvt(weight_hzb, whzb, (size_t)Ez * HHz,            stream);
    launch_cvt(weight_dzi, wdzi, (size_t)4 * Hz * Ez,         stream);
    launch_cvt(weight_dzH, wdzH, (size_t)4 * Hz * Ez,         stream);
    launch_cvt(weight_bz,  wbz,  (size_t)4 * Hz * Ez,         stream);

    // z embeddings (3072 waves)
    z_gemm<<<384, 256, 0, stream>>>(mtb, whzi, whzH, whzb, bias_i, bias_H,
                                    zib, zHb, zbb);

    // meta LSTM (512 blocks x 8 waves)
    meta_gemm_lstm<<<512, 256, 0, stream>>>(inb, mhb, mtb, wih, whh, bias_hyper,
                                            meta_c, out_metah, out_metac);

    // main LSTM (2048 blocks x 8 waves)
    main_gemm_lstm<<<2048, 256, 0, stream>>>(inb, mhb, wiH, wHH, zib, zHb, zbb,
                                             wdzi, wdzH, wbz, bias, main_c,
                                             out_mainh, out_mainc);
}
